// MambaBlock_59021440582435
// MI455X (gfx1250) — matrix operand order repaired and checked by reading, not compiled, not run on hardware
//
#include <hip/hip_runtime.h>
#include <hip/hip_bf16.h>

// ---------------------------------------------------------------------------
// Mamba block forward on gfx1250 (MI455X), wave32 + WMMA f16->f32.
// ---------------------------------------------------------------------------

typedef _Float16 v8h_t  __attribute__((ext_vector_type(8)));
typedef _Float16 v16h_t __attribute__((ext_vector_type(16)));
typedef float    v8f_t  __attribute__((ext_vector_type(8)));

// Problem constants
static constexpr int BATCH = 2;
static constexpr int SEQL  = 2048;
static constexpr int DM    = 1024;   // d_model
static constexpr int DI    = 2048;   // d_inner
static constexpr int DS    = 16;     // d_state
static constexpr int DTR   = 64;     // dt_rank
static constexpr int MROWS = BATCH * SEQL;          // 4096
static constexpr int NXZ   = 2 * DI;                // 4096
static constexpr int NXDBL = DTR + 2 * DS;          // 96

// Workspace layout (bytes)
static constexpr size_t O_XH    = 0;                                        // x f16        [M,DM]
static constexpr size_t O_WINH  = O_XH    + (size_t)MROWS * DM * 2;         // W_in f16     [NXZ,DM]
static constexpr size_t O_WXH   = O_WINH  + (size_t)NXZ   * DM * 2;         // W_xproj f16  [96,DI]
static constexpr size_t O_WDTH  = O_WXH   + (size_t)NXDBL * DI * 2;         // W_dt f16     [DI,DTR]
static constexpr size_t O_WOUTH = O_WDTH  + (size_t)DI    * DTR * 2;        // W_out f16    [DM,DI]
static constexpr size_t O_XZH   = O_WOUTH + (size_t)DM    * DI * 2;         // xz f16       [M,NXZ]
static constexpr size_t O_UH    = O_XZH   + (size_t)MROWS * NXZ * 2;        // u f16        [M,DI]
static constexpr size_t O_XDBL  = O_UH    + (size_t)MROWS * DI * 2;         // x_dbl f32    [M,96]
static constexpr size_t O_DTLH  = O_XDBL  + (size_t)MROWS * NXDBL * 4;      // dt_low f16   [M,DTR]
static constexpr size_t O_DELH  = O_DTLH  + (size_t)MROWS * DTR * 2;        // delta f16    [M,DI]
static constexpr size_t O_YH    = O_DELH  + (size_t)MROWS * DI * 2;         // y f16        [M,DI]

// ---------------------------------------------------------------------------
// f32 -> f16 convert
// ---------------------------------------------------------------------------
__global__ void cvt_f32_f16(const float* __restrict__ in,
                            _Float16* __restrict__ out, int n) {
    int i = blockIdx.x * blockDim.x + threadIdx.x;
    if (i < n) out[i] = (_Float16)in[i];
}

// Extract dt_low (first 64 cols of x_dbl) to f16 for the dt_proj WMMA GEMM
__global__ void extract_dtlow(const float* __restrict__ xdbl,
                              _Float16* __restrict__ dtl, int n) {
    int i = blockIdx.x * blockDim.x + threadIdx.x;
    if (i < n) {
        int m = i >> 6, r = i & 63;
        dtl[i] = (_Float16)xdbl[(size_t)m * NXDBL + r];
    }
}

// ---------------------------------------------------------------------------
// Generic WMMA GEMM:  out[M,N] = A[M,K](f16) * W[N,K](f16)^T  (+ epilogue)
//   EPI==0: plain store (OUT_T = float or _Float16)
//   EPI==1: softplus(val + bias[n])
// One wave computes a (16*TM) x (16*TN) tile; K stepped by 32.
// TM=TN=4 -> 16 WMMAs per 16 b128 loads per K-step (4x the round-1 reuse).
// EXEC must be all-ones for v_wmma: grids are exact multiples of 8 waves and
// the tail guard is wave-uniform.
// ---------------------------------------------------------------------------
template <int TM, int TN, int EPI, typename OUT_T>
__global__ void __launch_bounds__(256)
wmma_gemm(const _Float16* __restrict__ A,
          const _Float16* __restrict__ W,
          OUT_T* __restrict__ out,
          const float* __restrict__ bias,
          int M, int N, int K) {
    const int lane = threadIdx.x & 31;
    const int wave = blockIdx.x * (blockDim.x >> 5) + (threadIdx.x >> 5);
    const int rowGroups = M / (16 * TM);
    const int colGroups = N / (16 * TN);
    if (wave >= rowGroups * colGroups) return;   // wave-uniform

    const int m0 = (wave / colGroups) * (16 * TM);
    const int n0 = (wave % colGroups) * (16 * TN);

    const int half = lane >> 4;    // K-half select (contiguous k order, same in A and B)
    const int r    = lane & 15;    // row (A) / column (B/C/D)

    v8f_t acc[TM][TN] = {};

    // A: lane r holds row m0+i*16+r; elements 0..15 at K = k0+half*16 .. +15
    //    (contiguous, matching B)
    const _Float16* apb = A + (size_t)(m0 + r) * K + half * 16;
    // B: lane r holds column n0+t*16+r; K = k0 + half*16 + i -> one
    //    contiguous, 32B-aligned 32-byte load.
    const _Float16* bpb = W + (size_t)(n0 + r) * K + half * 16;

    for (int k0 = 0; k0 < K; k0 += 32) {
        __builtin_prefetch(apb + k0 + 64, 0, 1);   // global_prefetch_b8
        v16h_t a[TM];
#pragma unroll
        for (int i = 0; i < TM; ++i) {
            const _Float16* ap = apb + (size_t)i * 16 * K + k0;
            v8h_t lo = *(const v8h_t*)ap;
            v8h_t hi = *(const v8h_t*)(ap + 8);
#pragma unroll
            for (int q = 0; q < 8; ++q) { a[i][q] = lo[q]; a[i][q + 8] = hi[q]; }
        }
#pragma unroll
        for (int t = 0; t < TN; ++t) {
            v16h_t b = *(const v16h_t*)(bpb + (size_t)t * 16 * K + k0);
#pragma unroll
            for (int i = 0; i < TM; ++i)
                acc[i][t] = __builtin_amdgcn_wmma_f32_16x16x32_f16(
                    false, a[i], false, b, (short)0, acc[i][t], false, false);
        }
    }

    // C/D layout: VGPR rr -> row m0 + i*16 + half*8 + rr, col n0 + t*16 + r
#pragma unroll
    for (int i = 0; i < TM; ++i) {
#pragma unroll
        for (int t = 0; t < TN; ++t) {
            const int ncol = n0 + t * 16 + r;
#pragma unroll
            for (int rr = 0; rr < 8; ++rr) {
                const int mrow = m0 + i * 16 + half * 8 + rr;
                float v = acc[i][t][rr];
                if constexpr (EPI == 1) {
                    float xv = v + bias[ncol];
                    v = (xv > 20.0f) ? xv : log1pf(__expf(xv));
                }
                out[(size_t)mrow * N + ncol] = (OUT_T)v;
            }
        }
    }
}

// ---------------------------------------------------------------------------
// Causal depthwise conv1d (D_CONV=4) + bias + SiLU.  xz is [M, 4096] f16,
// x-branch occupies columns [0, DI).  u stored f16 [M, DI].
// ---------------------------------------------------------------------------
__global__ void conv_silu(const _Float16* __restrict__ xz,
                          const float* __restrict__ cw,
                          const float* __restrict__ cb,
                          _Float16* __restrict__ u) {
    size_t i = (size_t)blockIdx.x * blockDim.x + threadIdx.x; // over M*DI
    if (i >= (size_t)MROWS * DI) return;
    const int d = (int)(i % DI);
    const size_t m = i / DI;
    const int l = (int)(m % SEQL);

    float acc = cb[d];
#pragma unroll
    for (int j = 0; j < 4; ++j) {
        int lj = l - 3 + j;
        if (lj >= 0)
            acc += (float)xz[(m - 3 + j) * NXZ + d] * cw[d * 4 + j];
    }
    float s = acc / (1.0f + __expf(-acc));      // SiLU
    u[i] = (_Float16)s;
}

// ---------------------------------------------------------------------------
// Sequential selective-state-space scan + skip + gate.
// One thread per (batch, d_inner) channel; 16 states in registers.
// All 256 threads of a block share the same batch index, so B_t/C_t (32
// floats, uniform across the block) are staged through LDS in chunks of
// TCH timesteps: one coalesced global load per chunk + ds broadcasts,
// instead of 32 per-lane gathers per step.
// ---------------------------------------------------------------------------
static constexpr int TCH = 16;   // timesteps staged per LDS chunk

__global__ void ssm_scan_gate(const _Float16* __restrict__ delta_h,
                              const _Float16* __restrict__ u_h,
                              const _Float16* __restrict__ xz_h,
                              const float* __restrict__ xdbl,
                              const float* __restrict__ A_log,
                              const float* __restrict__ Dskip,
                              _Float16* __restrict__ y_h) {
    __shared__ float sBC[TCH * 2 * DS];          // 16 steps x (B[16]|C[16])
    const int tid = threadIdx.x;
    const int gid = blockIdx.x * blockDim.x + tid;
    const int b = gid / DI;                      // uniform per block
    const int d = gid % DI;

    float a[DS], h[DS];
#pragma unroll
    for (int n = 0; n < DS; ++n) {
        a[n] = -__expf(A_log[(size_t)d * DS + n]);
        h[n] = 0.0f;
    }
    const float dsk = Dskip[d];

    for (int l0 = 0; l0 < SEQL; l0 += TCH) {
        // cooperative stage of B/C for TCH timesteps (512 floats, 2 per thread)
        for (int i = tid; i < TCH * 2 * DS; i += 256) {
            const int t = i >> 5, j = i & 31;
            sBC[i] = xdbl[((size_t)b * SEQL + l0 + t) * NXDBL + DTR + j];
        }
        __syncthreads();

        for (int tt = 0; tt < TCH; ++tt) {
            const size_t m  = (size_t)b * SEQL + l0 + tt;
            const float dt  = (float)delta_h[m * DI + d];
            const float ut  = (float)u_h[m * DI + d];
            const float du  = dt * ut;
            const float* Bp = &sBC[tt * 32];
            const float* Cp = &sBC[tt * 32 + DS];

            float y = 0.0f;
#pragma unroll
            for (int n = 0; n < DS; ++n) {
                h[n] = __expf(dt * a[n]) * h[n] + du * Bp[n];
                y += h[n] * Cp[n];
            }
            const float zt = (float)xz_h[m * NXZ + DI + d]; // gate z
            const float g  = zt / (1.0f + __expf(-zt));     // SiLU(z)
            y_h[m * DI + d] = (_Float16)((y + ut * dsk) * g);
        }
        __syncthreads();
    }
}

// ---------------------------------------------------------------------------
// Launcher
// ---------------------------------------------------------------------------
extern "C" void kernel_launch(void* const* d_in, const int* in_sizes, int n_in,
                              void* d_out, int out_size, void* d_ws, size_t ws_size,
                              hipStream_t stream) {
    (void)in_sizes; (void)n_in; (void)out_size; (void)ws_size;
    const float* x      = (const float*)d_in[0];
    const float* W_in   = (const float*)d_in[1];
    const float* conv_w = (const float*)d_in[2];
    const float* conv_b = (const float*)d_in[3];
    const float* W_xp   = (const float*)d_in[4];
    const float* W_dt   = (const float*)d_in[5];
    const float* b_dt   = (const float*)d_in[6];
    const float* A_log  = (const float*)d_in[7];
    const float* Dskip  = (const float*)d_in[8];
    const float* W_out  = (const float*)d_in[9];
    float* out = (float*)d_out;

    char* ws = (char*)d_ws;
    _Float16* xh    = (_Float16*)(ws + O_XH);
    _Float16* Winh  = (_Float16*)(ws + O_WINH);
    _Float16* Wxh   = (_Float16*)(ws + O_WXH);
    _Float16* Wdth  = (_Float16*)(ws + O_WDTH);
    _Float16* Wouth = (_Float16*)(ws + O_WOUTH);
    _Float16* xzh   = (_Float16*)(ws + O_XZH);
    _Float16* uh    = (_Float16*)(ws + O_UH);
    float*    xdbl  = (float*)   (ws + O_XDBL);
    _Float16* dtlh  = (_Float16*)(ws + O_DTLH);
    _Float16* delh  = (_Float16*)(ws + O_DELH);
    _Float16* yh    = (_Float16*)(ws + O_YH);

    // 1) f32 -> f16 converts (weights resident in 192MB L2 thereafter)
    cvt_f32_f16<<<(MROWS * DM) / 256, 256, 0, stream>>>(x, xh, MROWS * DM);
    cvt_f32_f16<<<(NXZ * DM) / 256, 256, 0, stream>>>(W_in, Winh, NXZ * DM);
    cvt_f32_f16<<<(NXDBL * DI) / 256, 256, 0, stream>>>(W_xp, Wxh, NXDBL * DI);
    cvt_f32_f16<<<(DI * DTR) / 256, 256, 0, stream>>>(W_dt, Wdth, DI * DTR);
    cvt_f32_f16<<<(DM * DI) / 256, 256, 0, stream>>>(W_out, Wouth, DM * DI);

    // 2) in_proj: xz[M,4096] = x * W_in^T   (64x64 rowxcol groups -> 512 blocks)
    wmma_gemm<4, 4, 0, _Float16><<<512, 256, 0, stream>>>(
        xh, Winh, xzh, nullptr, MROWS, NXZ, DM);

    // 3) causal depthwise conv + SiLU -> u
    conv_silu<<<(MROWS * DI) / 256, 256, 0, stream>>>(xzh, conv_w, conv_b, uh);

    // 4) x_proj: x_dbl[M,96] = u * W_xproj^T  (64x3 groups -> 24 blocks)
    wmma_gemm<4, 2, 0, float><<<24, 256, 0, stream>>>(
        uh, Wxh, xdbl, nullptr, MROWS, NXDBL, DI);

    // 5) dt_low slice to f16
    extract_dtlow<<<(MROWS * DTR) / 256, 256, 0, stream>>>(xdbl, dtlh, MROWS * DTR);

    // 6) dt_proj + softplus: delta[M,2048]  (64x32 groups -> 256 blocks)
    wmma_gemm<4, 4, 1, _Float16><<<256, 256, 0, stream>>>(
        dtlh, Wdth, delh, b_dt, MROWS, DI, DTR);

    // 7) sequential SSM scan + skip + SiLU gate -> y (4096 channels)
    ssm_scan_gate<<<(BATCH * DI) / 256, 256, 0, stream>>>(
        delh, uh, xzh, xdbl, A_log, Dskip, yh);

    // 8) out_proj: out[M,1024] = y * W_out^T  (64x16 groups -> 128 blocks)
    wmma_gemm<4, 4, 0, float><<<128, 256, 0, stream>>>(
        yh, Wouth, out, nullptr, MROWS, DM, DI);
}
